// DSConv_10290741641256
// MI455X (gfx1250) — matrix operand order repaired and checked by reading, not compiled, not run on hardware
//
#include <hip/hip_runtime.h>

#define BB 4
#define CC 32
#define OO 32
#define KK 9
#define HH 256
#define WW 256
#define HW (HH*WW)
#define EPSBN 1e-5f

typedef __attribute__((ext_vector_type(16))) _Float16 v16h;
typedef __attribute__((ext_vector_type(8)))  float    v8f;

// ---------------------------------------------------------------------------
// Setup: combined weights  Wc_i = fus_w[:, i*32:(i+1)*32] @ W_i  (i = std, x, y)
// written directly in a per-lane WMMA A 16x32 layout with contiguous k (c = 16*(L/16) + e):
//   idx = mode*9216 + ((k*2 + t)*32 + L)*16 + e
// ---------------------------------------------------------------------------
__global__ void k_combine(const float* __restrict__ conv_w, const float* __restrict__ dscx_w,
                          const float* __restrict__ dscy_w, const float* __restrict__ fus_w,
                          _Float16* __restrict__ wtile) {
    int idx = blockIdx.x * blockDim.x + threadIdx.x;     // 27648 entries
    if (idx >= 3 * KK * 2 * 32 * 16) return;
    int e = idx & 15, L = (idx >> 4) & 31, t = (idx >> 9) & 1;
    int kk = idx >> 10;                  // 0..26
    int mode = kk / KK, k = kk % KK;
    int hiL = L >> 4;
    int c = e + hiL * 16;   // A-layout K index
    int o = t * 16 + (L & 15);                                    // A-layout M index
    const float* Wm = (mode == 0) ? conv_w : (mode == 1 ? dscx_w : dscy_w);
    const float* fr = fus_w + o * 96 + mode * 32;
    float s = 0.f;
    for (int m = 0; m < 32; m++) s += fr[m] * Wm[(m * CC + c) * KK + k];
    wtile[idx] = (_Float16)s;
}

// fused bias contribution:  biasf[o] = sum_c fus_w[o, c] * conv_b[c]
__global__ void k_bias(const float* __restrict__ fus_w, const float* __restrict__ conv_b,
                       float* __restrict__ biasf) {
    int o = threadIdx.x;
    float s = 0.f;
    for (int c = 0; c < CC; c++) s += fus_w[o * 96 + c] * conv_b[c];
    biasf[o] = s;
}

// ---------------------------------------------------------------------------
// Pass 1 of offset-branch BN: depthwise 3x3 conv, accumulate sum/sumsq only.
// One block = one (b,c) row of 256 pixels.
// ---------------------------------------------------------------------------
__global__ void k_dw_stats(const float* __restrict__ x, const float* __restrict__ dww,
                           const float* __restrict__ dwb, float* __restrict__ sums) {
    int w  = threadIdx.x;
    int h  = blockIdx.x;
    int bc = blockIdx.y;
    int c  = bc & (CC - 1);
    const float* xc = x + (size_t)bc * HW;
    float acc = dwb[c];
#pragma unroll
    for (int ky = 0; ky < 3; ky++)
#pragma unroll
        for (int kx = 0; kx < 3; kx++) {
            int yy = h + ky - 1, xx = w + kx - 1;
            if (yy >= 0 && yy < HH && xx >= 0 && xx < WW)
                acc += dww[c * 9 + ky * 3 + kx] * xc[yy * WW + xx];
        }
    __shared__ float s1[256], s2[256];
    s1[w] = acc; s2[w] = acc * acc;
    __syncthreads();
    for (int s = 128; s > 0; s >>= 1) {
        if (w < s) { s1[w] += s1[w + s]; s2[w] += s2[w + s]; }
        __syncthreads();
    }
    if (w == 0) {
        atomicAdd(&sums[c],      s1[0]);
        atomicAdd(&sums[CC + c], s2[0]);
    }
}

// BN stats -> per-channel scale/shift.
__global__ void k_finalize(const float* __restrict__ sums, const float* __restrict__ g,
                           const float* __restrict__ be, float* __restrict__ scsh,
                           float invN) {
    int c = threadIdx.x;
    float mean = sums[c] * invN;
    float var  = sums[CC + c] * invN - mean * mean;
    float sc   = g[c] * rsqrtf(var + EPSBN);
    scsh[c]      = sc;
    scsh[CC + c] = be[c] - mean * sc;
}

// ---------------------------------------------------------------------------
// Pass 2: recompute depthwise conv, BN+ReLU, pointwise C->K => offsets.
// One block = one (b,h) row.  (Recompute beats a 32MB store+load round trip.)
// ---------------------------------------------------------------------------
__global__ void k_pw(const float* __restrict__ x, const float* __restrict__ dww,
                     const float* __restrict__ dwb, const float* __restrict__ scsh,
                     const float* __restrict__ pww, const float* __restrict__ pwb,
                     float* __restrict__ off) {
    __shared__ float sdw[CC * 9], sdb[CC], spw[KK * CC], spb[KK], ssc[2 * CC];
    for (int i = threadIdx.x; i < CC * 9; i += blockDim.x) { sdw[i] = dww[i]; spw[i] = pww[i]; }
    if (threadIdx.x < CC)     sdb[threadIdx.x] = dwb[threadIdx.x];
    if (threadIdx.x < KK)     spb[threadIdx.x] = pwb[threadIdx.x];
    if (threadIdx.x < 2 * CC) ssc[threadIdx.x] = scsh[threadIdx.x];
    __syncthreads();

    int w = threadIdx.x, h = blockIdx.x, b = blockIdx.y;
    float acc[KK];
#pragma unroll
    for (int k = 0; k < KK; k++) acc[k] = spb[k];
    for (int c = 0; c < CC; c++) {
        const float* xc = x + (size_t)(b * CC + c) * HW;
        float v = sdb[c];
#pragma unroll
        for (int ky = 0; ky < 3; ky++)
#pragma unroll
            for (int kx = 0; kx < 3; kx++) {
                int yy = h + ky - 1, xx = w + kx - 1;
                if (yy >= 0 && yy < HH && xx >= 0 && xx < WW)
                    v += sdw[c * 9 + ky * 3 + kx] * xc[yy * WW + xx];
            }
        v = fmaxf(v * ssc[c] + ssc[CC + c], 0.f);
#pragma unroll
        for (int k = 0; k < KK; k++) acc[k] += spw[k * CC + c] * v;
    }
#pragma unroll
    for (int k = 0; k < KK; k++)
        off[((size_t)(b * KK + k) * HH + h) * WW + w] = acc[k];
}

// ---------------------------------------------------------------------------
// Main fused kernel: 27 WMMA K-chunks (3 branches x 9 taps) with the
// fuse-premultiplied weights, accumulating the pre-BN fused output directly.
// Tap loops deliberately NOT unrolled (pressure); gathers issued as a 32-load
// clause into register arrays before the lerp/convert phase.
// ---------------------------------------------------------------------------
__global__ void k_wmma_main(const float* __restrict__ x, const _Float16* __restrict__ wtile,
                            const float* __restrict__ biasf, const float* __restrict__ off_x,
                            const float* __restrict__ off_y, float* __restrict__ fout,
                            float* __restrict__ sums) {
    __shared__ _Float16 __attribute__((aligned(32))) aLds[KK * 2 * 32 * 16];
    __shared__ float ssum[OO], ssq[OO];
    if (threadIdx.x < OO) { ssum[threadIdx.x] = 0.f; ssq[threadIdx.x] = 0.f; }

    int lane = threadIdx.x & 31, wave = threadIdx.x >> 5;
    int hi = lane >> 4;
    int wp = blockIdx.x * 128 + wave * 16 + (lane & 15);  // pixel column (B-matrix N)
    int h  = blockIdx.y, b = blockIdx.z;
    int cg = hi * 16;                                     // channel group (B-matrix K rows)
    const v16h* aT = (const v16h*)aLds;
    // lane's channel-group base into x: x[b, cg, 0, 0]
    const float* xg = x + (size_t)(b * CC + cg) * HW;

    v8f acc0 = {}, acc1 = {};
    for (int mode = 0; mode < 3; mode++) {
        __syncthreads();
        {   // stage this mode's 18KB of A-tiles into LDS (coalesced 16B copies)
            const float4* src = (const float4*)(wtile + (size_t)mode * (KK * 2 * 32 * 16));
            float4* dst = (float4*)aLds;
            for (int i = threadIdx.x; i < KK * 2 * 32 * 16 / 8; i += blockDim.x) dst[i] = src[i];
        }
        __syncthreads();

        if (mode == 0) {
#pragma unroll 1
            for (int k = 0; k < KK; k++) {
                int yy = h + k / 3 - 1, xx = wp + (k % 3) - 1;
                bool vv = (yy >= 0) & (yy < HH) & (xx >= 0) & (xx < WW);
                int cy = vv ? yy : 0, cx = vv ? xx : 0;
                float wv = vv ? 1.f : 0.f;
                const float* base = xg + cy * WW + cx;
                float g[16];
#pragma unroll
                for (int e = 0; e < 16; e++) g[e] = base[e * HW];     // 16-load clause
                v16h bv;
#pragma unroll
                for (int e = 0; e < 16; e++) bv[e] = (_Float16)(wv * g[e]);
                v16h alo = aT[(k * 2 + 0) * 32 + lane];
                v16h ahi = aT[(k * 2 + 1) * 32 + lane];
                acc0 = __builtin_amdgcn_wmma_f32_16x16x32_f16(false, alo, false, bv, (short)0, acc0, false, false);
                acc1 = __builtin_amdgcn_wmma_f32_16x16x32_f16(false, ahi, false, bv, (short)0, acc1, false, false);
            }
        } else if (mode == 1) {
#pragma unroll 1
            for (int k = 0; k < KK; k++) {
                float off = off_x[((size_t)(b * KK + k) * HH + h) * WW + wp];
                float fx  = (float)wp + off;
                float x0f = floorf(fx);
                int ix0 = (int)x0f, ix1 = ix0 + 1;
                float f  = fx - x0f;
                float w0 = (1.f - f) * ((ix0 >= 0 && ix0 < WW) ? 1.f : 0.f);
                float w1 = f * ((ix1 >= 0 && ix1 < WW) ? 1.f : 0.f);
                int cx0 = min(max(ix0, 0), WW - 1), cx1 = min(max(ix1, 0), WW - 1);
                const float* b0p = xg + h * WW + cx0;
                const float* b1p = xg + h * WW + cx1;
                float g0[16], g1[16];
#pragma unroll
                for (int e = 0; e < 16; e++) { g0[e] = b0p[e * HW]; g1[e] = b1p[e * HW]; } // 32-load clause
                v16h bv;
#pragma unroll
                for (int e = 0; e < 16; e++) bv[e] = (_Float16)(w0 * g0[e] + w1 * g1[e]);
                v16h alo = aT[(k * 2 + 0) * 32 + lane];
                v16h ahi = aT[(k * 2 + 1) * 32 + lane];
                acc0 = __builtin_amdgcn_wmma_f32_16x16x32_f16(false, alo, false, bv, (short)0, acc0, false, false);
                acc1 = __builtin_amdgcn_wmma_f32_16x16x32_f16(false, ahi, false, bv, (short)0, acc1, false, false);
            }
        } else {
#pragma unroll 1
            for (int k = 0; k < KK; k++) {
                float off = off_y[((size_t)(b * KK + k) * HH + h) * WW + wp];
                float fy  = (float)h + off;
                float y0f = floorf(fy);
                int iy0 = (int)y0f, iy1 = iy0 + 1;
                float f  = fy - y0f;
                float w0 = (1.f - f) * ((iy0 >= 0 && iy0 < HH) ? 1.f : 0.f);
                float w1 = f * ((iy1 >= 0 && iy1 < HH) ? 1.f : 0.f);
                int cy0 = min(max(iy0, 0), HH - 1), cy1 = min(max(iy1, 0), HH - 1);
                const float* b0p = xg + cy0 * WW + wp;
                const float* b1p = xg + cy1 * WW + wp;
                float g0[16], g1[16];
#pragma unroll
                for (int e = 0; e < 16; e++) { g0[e] = b0p[e * HW]; g1[e] = b1p[e * HW]; } // 32-load clause
                v16h bv;
#pragma unroll
                for (int e = 0; e < 16; e++) bv[e] = (_Float16)(w0 * g0[e] + w1 * g1[e]);
                v16h alo = aT[(k * 2 + 0) * 32 + lane];
                v16h ahi = aT[(k * 2 + 1) * 32 + lane];
                acc0 = __builtin_amdgcn_wmma_f32_16x16x32_f16(false, alo, false, bv, (short)0, acc0, false, false);
                acc1 = __builtin_amdgcn_wmma_f32_16x16x32_f16(false, ahi, false, bv, (short)0, acc1, false, false);
            }
        }
    }

#pragma unroll
    for (int r = 0; r < 8; r++) {
        int m = r + hi * 8;                   // C/D layout: VGPR r -> M = r (+8 for lanes 16-31)
        float v0 = acc0[r] + biasf[m];
        float v1 = acc1[r] + biasf[16 + m];
        fout[((size_t)(b * OO + m)      * HH + h) * WW + wp] = v0;
        fout[((size_t)(b * OO + 16 + m) * HH + h) * WW + wp] = v1;
        float s0 = v0, q0 = v0 * v0, s1 = v1, q1 = v1 * v1;
#pragma unroll
        for (int msk = 1; msk < 16; msk <<= 1) {
            s0 += __shfl_xor(s0, msk, 16);
            q0 += __shfl_xor(q0, msk, 16);
            s1 += __shfl_xor(s1, msk, 16);
            q1 += __shfl_xor(q1, msk, 16);
        }
        if ((lane & 15) == 0) {
            atomicAdd(&ssum[m], s0);       atomicAdd(&ssq[m], q0);
            atomicAdd(&ssum[16 + m], s1);  atomicAdd(&ssq[16 + m], q1);
        }
    }
    __syncthreads();
    if (threadIdx.x < OO) {
        atomicAdd(&sums[threadIdx.x],      ssum[threadIdx.x]);
        atomicAdd(&sums[OO + threadIdx.x], ssq[threadIdx.x]);
    }
}

// ---------------------------------------------------------------------------
// Final BN + ReLU (elementwise).  H*W == 1<<16.
// ---------------------------------------------------------------------------
__global__ void k_bn_relu(const float* __restrict__ fin, const float* __restrict__ scsh,
                          float* __restrict__ out) {
    size_t i = (size_t)blockIdx.x * blockDim.x + threadIdx.x;
    int c = (int)((i >> 16) & (OO - 1));
    out[i] = fmaxf(fin[i] * scsh[c] + scsh[OO + c], 0.f);
}

// ---------------------------------------------------------------------------
extern "C" void kernel_launch(void* const* d_in, const int* in_sizes, int n_in,
                              void* d_out, int out_size, void* d_ws, size_t ws_size,
                              hipStream_t stream) {
    const float* x       = (const float*)d_in[0];
    const float* conv_w  = (const float*)d_in[1];
    const float* conv_b  = (const float*)d_in[2];
    const float* ox_dww  = (const float*)d_in[3];
    const float* ox_dwb  = (const float*)d_in[4];
    const float* ox_bng  = (const float*)d_in[5];
    const float* ox_bnb  = (const float*)d_in[6];
    const float* ox_pww  = (const float*)d_in[7];
    const float* ox_pwb  = (const float*)d_in[8];
    const float* oy_dww  = (const float*)d_in[9];
    const float* oy_dwb  = (const float*)d_in[10];
    const float* oy_bng  = (const float*)d_in[11];
    const float* oy_bnb  = (const float*)d_in[12];
    const float* oy_pww  = (const float*)d_in[13];
    const float* oy_pwb  = (const float*)d_in[14];
    const float* dscx_w  = (const float*)d_in[15];
    const float* dscy_w  = (const float*)d_in[16];
    const float* fus_w   = (const float*)d_in[17];
    const float* fus_bng = (const float*)d_in[18];
    const float* fus_bnb = (const float*)d_in[19];

    char* ws = (char*)d_ws;
    float*     off_x = (float*)(ws);                            //  9.44 MB
    float*     off_y = (float*)(ws + (size_t)9437184);          //  9.44 MB
    float*     fout  = (float*)(ws + (size_t)18874368);         // 32    MB pre-BN fused
    _Float16*  wtile = (_Float16*)(ws + (size_t)52428800);      // 55296 B combined A-tiles
    float*     biasf = (float*)(ws + (size_t)52484096);         // 128 B
    float*     stats = (float*)(ws + (size_t)52484224);         // 384 floats

    hipMemsetAsync(stats, 0, 384 * sizeof(float), stream);
    float invN = 1.f / (float)(BB * HH * WW);
    dim3 blk(256);

    // setup: fuse-premultiplied WMMA A-tiles + folded bias
    k_combine<<<108, blk, 0, stream>>>(conv_w, dscx_w, dscy_w, fus_w, wtile);
    k_bias<<<1, OO, 0, stream>>>(fus_w, conv_b, biasf);

    // offset branch x (stats pass, then recompute+BN+ReLU+pointwise)
    k_dw_stats<<<dim3(HH, BB * CC), blk, 0, stream>>>(x, ox_dww, ox_dwb, stats);
    k_finalize<<<1, CC, 0, stream>>>(stats, ox_bng, ox_bnb, stats + 64, invN);
    k_pw<<<dim3(HH, BB), blk, 0, stream>>>(x, ox_dww, ox_dwb, stats + 64, ox_pww, ox_pwb, off_x);

    // offset branch y
    k_dw_stats<<<dim3(HH, BB * CC), blk, 0, stream>>>(x, oy_dww, oy_dwb, stats + 128);
    k_finalize<<<1, CC, 0, stream>>>(stats + 128, oy_bng, oy_bnb, stats + 192, invN);
    k_pw<<<dim3(HH, BB), blk, 0, stream>>>(x, oy_dww, oy_dwb, stats + 192, oy_pww, oy_pwb, off_y);

    // single fused WMMA pass over all three branches (27 K-chunks)
    k_wmma_main<<<dim3(WW / 128, HH, BB), blk, 0, stream>>>(x, wtile, biasf, off_x, off_y,
                                                            fout, stats + 256);
    k_finalize<<<1, CC, 0, stream>>>(stats + 256, fus_bng, fus_bnb, stats + 320, invN);
    k_bn_relu<<<(BB * OO * HW) / 256, blk, 0, stream>>>(fout, stats + 320, (float*)d_out);
}
